// SwinTransformerBlock_17858474016992
// MI455X (gfx1250) — matrix operand order repaired and checked by reading, not compiled, not run on hardware
//
#include <hip/hip_runtime.h>
#include <hip/hip_bf16.h>

// ---------------------------------------------------------------------------
// Swin Transformer block for MI455X (gfx1250, wave32, WMMA).
// All GEMMs run through v_wmma_f32_16x16x32_f16 (f16 in, f32 accum).
// Activations kept in f16 to halve HBM traffic (23.3 TB/s budget ~26us for
// ~600MB of intermediates; every intermediate also fits the 192MB L2, so
// GEMM re-reads of A hit L2, not HBM). Weights converted once to f16
// row-major [N,K] so B-fragments are contiguous 16-half loads per lane.
// GEMM wave tile = 32(M) x 64(N): 8 WMMAs per k=32 step from 2 A-frags +
// 4 B-frags (back-to-back WMMA issue for the XDL pipes), with
// global_prefetch of the next A k-slice.
// ---------------------------------------------------------------------------

#define DIM 192
#define HEADS 6
#define HDIM 32
#define WSZ 7
#define NWT 49          // tokens per window
#define SHIFT 3
#define HID 768
#define HH 56
#define NTOK 100352     // 32 * 56 * 56
#define NWIN 2048       // 32 * 8 * 8
#define QK_SCALE 0.17677669529663687f   // 32^-0.5
#define LN_EPS 1e-5f

typedef __attribute__((ext_vector_type(16))) _Float16 v16h;
typedef __attribute__((ext_vector_type(8)))  _Float16 v8h;
typedef __attribute__((ext_vector_type(8)))  float    v8f;

// ---- WMMA fragment loaders (contiguous k order, same in A and B, wave32) -------------------
// A 16x32 f16: lane = M (mod 16); halfs[i]=A[m][k0+16*hi+i] (contiguous)
__device__ __forceinline__ v16h load_a_frag(const _Float16* A, int lda,
                                            int m0, int k0, int lane) {
  const int r  = lane & 15;
  const int hi = lane >> 4;
  const _Float16* p = A + (size_t)(m0 + r) * lda + k0 + hi * 16;
  union { v16h v; v8h h[2]; } u;
  u.h[0] = *(const v8h*)(p);
  u.h[1] = *(const v8h*)(p + 8);
  return u.v;
}

// B 32x16 f16 with B[k][n] = Wt[n][k] (Wt row-major [N,K]):
// lane = N (mod 16); halfs[i] = Wt[n][k0 + 16*hi + i]  (contiguous!)
__device__ __forceinline__ v16h load_b_frag(const _Float16* Wt, int ldw,
                                            int n0, int k0, int lane) {
  const int n  = n0 + (lane & 15);
  const int kb = k0 + (lane >> 4) * 16;
  const _Float16* p = Wt + (size_t)n * ldw + kb;
  union { v16h v; v8h h[2]; } u;
  u.h[0] = *(const v8h*)(p);
  u.h[1] = *(const v8h*)(p + 8);
  return u.v;
}

// ---- small elementwise helpers --------------------------------------------
__global__ __launch_bounds__(256) void f32_to_f16_kernel(
    const float* __restrict__ in, _Float16* __restrict__ out, int n) {
  int i = blockIdx.x * 256 + threadIdx.x;
  if (i < n) out[i] = (_Float16)in[i];
}

// LayerNorm over C=192, one wave per token, output f16.
__global__ __launch_bounds__(256) void layernorm_f16_kernel(
    const float* __restrict__ x, const float* __restrict__ w,
    const float* __restrict__ b, _Float16* __restrict__ out, int ntok) {
  const int tok  = blockIdx.x * 8 + (threadIdx.x >> 5);
  const int lane = threadIdx.x & 31;
  if (tok >= ntok) return;
  const float* xp = x + (size_t)tok * DIM;
  float vals[6];
  float s = 0.f;
  #pragma unroll
  for (int i = 0; i < 6; ++i) { vals[i] = xp[lane + 32 * i]; s += vals[i]; }
  #pragma unroll
  for (int off = 16; off; off >>= 1) s += __shfl_xor(s, off, 32);
  const float mu = s * (1.0f / DIM);
  float var = 0.f;
  #pragma unroll
  for (int i = 0; i < 6; ++i) { float d = vals[i] - mu; var += d * d; }
  #pragma unroll
  for (int off = 16; off; off >>= 1) var += __shfl_xor(var, off, 32);
  const float rstd = rsqrtf(var * (1.0f / DIM) + LN_EPS);
  _Float16* op = out + (size_t)tok * DIM;
  #pragma unroll
  for (int i = 0; i < 6; ++i) {
    const int c = lane + 32 * i;
    op[c] = (_Float16)((vals[i] - mu) * rstd * w[c] + b[c]);
  }
}

// ---- generic WMMA GEMM: out[M,N] = A[M,K](f16) @ Wt[N,K]^T + bias ----------
// Wave tile 32(M) x 64(N); block = 8 waves = 256 rows x 64 cols.
enum { EPI_NONE = 0, EPI_GELU = 1 };

template <typename OutT, int EPI, bool RESID>
__global__ __launch_bounds__(256) void wmma_gemm_kernel(
    const _Float16* __restrict__ A, const _Float16* __restrict__ Wt,
    const float* __restrict__ bias, const float* __restrict__ resid,
    OutT* __restrict__ out, int M, int N, int K) {
  const int lane = threadIdx.x & 31;
  const int wave = threadIdx.x >> 5;
  const int m0 = (blockIdx.x * 8 + wave) * 32;
  const int n0 = blockIdx.y * 64;
  if (m0 >= M) return;

  v8f acc[2][4] = {};
  for (int k0 = 0; k0 < K; k0 += 32) {
    if (k0 + 32 < K) {
      // prefetch next A k-slice (global_prefetch_b8); weights are L2-resident
      const _Float16* pf = A + (size_t)(m0 + (lane & 15)) * K + k0 + 32;
      __builtin_prefetch(pf, 0, 0);
      __builtin_prefetch(pf + (size_t)16 * K, 0, 0);
    }
    const v16h a0 = load_a_frag(A, K, m0, k0, lane);
    const v16h a1 = load_a_frag(A, K, m0 + 16, k0, lane);
    #pragma unroll
    for (int t = 0; t < 4; ++t) {
      const v16h b = load_b_frag(Wt, K, n0 + t * 16, k0, lane);
      acc[0][t] = __builtin_amdgcn_wmma_f32_16x16x32_f16(
          false, a0, false, b, (short)0, acc[0][t], false, false);
      acc[1][t] = __builtin_amdgcn_wmma_f32_16x16x32_f16(
          false, a1, false, b, (short)0, acc[1][t], false, false);
    }
  }

  const int nl = lane & 15;
  const int hi = lane >> 4;
  #pragma unroll
  for (int mt = 0; mt < 2; ++mt) {
    #pragma unroll
    for (int t = 0; t < 4; ++t) {
      const int n = n0 + t * 16 + nl;
      const float bv = bias[n];
      #pragma unroll
      for (int v = 0; v < 8; ++v) {
        const int m = m0 + mt * 16 + v + 8 * hi;
        float val = acc[mt][t][v] + bv;
        if (EPI == EPI_GELU)
          val = 0.5f * val * (1.0f + erff(val * 0.70710678118654752f));
        if (RESID) val += resid[(size_t)m * N + n];
        out[(size_t)m * N + n] = (OutT)val;
      }
    }
  }
}

// ---- windowed attention ----------------------------------------------------
// shifted-window token r of window widx <-> global token index (gather AND
// scatter use the same map: roll(-S) then partition == reverse of
// reverse-partition then roll(+S)).
__device__ __forceinline__ int win_tok_to_global(int widx, int r) {
  const int b  = widx >> 6;
  const int wi = widx & 63;
  const int wy = wi >> 3, wx = wi & 7;
  const int iy = r / WSZ, ix = r % WSZ;
  const int y = (wy * WSZ + iy + SHIFT) % HH;
  const int x = (wx * WSZ + ix + SHIFT) % HH;
  return b * (HH * HH) + y * HH + x;
}

// grid = (NWIN, HEADS); 128 threads = 4 waves; wave w owns query rows 16w..16w+15
// of the 64-row padded tile (rows >= 49 are zero / masked).
__global__ __launch_bounds__(128) void window_attention_kernel(
    const _Float16* __restrict__ qkv, const float* __restrict__ rpb,
    _Float16* __restrict__ attnout) {
  const int widx = blockIdx.x;
  const int h    = blockIdx.y;
  const int tid  = threadIdx.x;
  const int lane = tid & 31;
  const int wave = tid >> 5;

  __shared__ __align__(16) _Float16 Qs[64 * 32];   // [row][d], q pre-scaled
  __shared__ __align__(16) _Float16 Ks[64 * 32];   // [row][d]
  __shared__ __align__(16) _Float16 Vt[32 * 64];   // [d][row] (transposed)
  __shared__ __align__(16) _Float16 Ps[64 * 64];   // softmax probs [row][col]

  // gather q/k/v for this (window, head): 64 rows x 2 half-rows x 3 mats
  for (int idx = tid; idx < 384; idx += 128) {
    const int mat  = idx >> 7;
    const int rem  = idx & 127;
    const int r    = rem >> 1;
    const int cb   = (rem & 1) * 16;
    if (r < NWT) {
      const int t = win_tok_to_global(widx, r);
      const _Float16* src = qkv + (size_t)t * (3 * DIM) + mat * DIM + h * HDIM + cb;
      if (mat == 0) {
        #pragma unroll
        for (int i = 0; i < 16; ++i)
          Qs[r * 32 + cb + i] = (_Float16)((float)src[i] * QK_SCALE);
      } else if (mat == 1) {
        #pragma unroll
        for (int i = 0; i < 16; ++i) Ks[r * 32 + cb + i] = src[i];
      } else {
        #pragma unroll
        for (int i = 0; i < 16; ++i) Vt[(cb + i) * 64 + r] = src[i];
      }
    } else {
      if (mat == 0) {
        #pragma unroll
        for (int i = 0; i < 16; ++i) Qs[r * 32 + cb + i] = (_Float16)0.f;
      } else if (mat == 1) {
        #pragma unroll
        for (int i = 0; i < 16; ++i) Ks[r * 32 + cb + i] = (_Float16)0.f;
      } else {
        #pragma unroll
        for (int i = 0; i < 16; ++i) Vt[(cb + i) * 64 + r] = (_Float16)0.f;
      }
    }
  }
  __syncthreads();

  // S = (Q*scale) @ K^T : 1x4 tiles per wave, K-dim = d = 32 (one WMMA each)
  v8f s[4] = {};
  {
    const v16h qa = load_a_frag(Qs, 32, wave * 16, 0, lane);
    #pragma unroll
    for (int c = 0; c < 4; ++c) {
      const v16h kb = load_b_frag(Ks, 32, c * 16, 0, lane);
      s[c] = __builtin_amdgcn_wmma_f32_16x16x32_f16(
          false, qa, false, kb, (short)0, s[c], false, false);
    }
  }

  // bias + mask + softmax. C layout: lane nl = col, VGPR v <-> row v + 8*hi.
  const int nl = lane & 15;
  const int hi = lane >> 4;
  #pragma unroll
  for (int v = 0; v < 8; ++v) {
    const int i  = wave * 16 + v + 8 * hi;         // query token in window
    const int ii = (i < NWT) ? i : 0;              // clamp (row discarded)
    const int iy = ii / WSZ, ix = ii % WSZ;
    float vals[4];
    float rowmax = -1e30f;
    #pragma unroll
    for (int c = 0; c < 4; ++c) {
      const int j = c * 16 + nl;                   // key token in window
      float sv = s[c][v];
      if (j < NWT) {
        const int jy = j / WSZ, jx = j % WSZ;
        const int ridx = (iy - jy + WSZ - 1) * (2 * WSZ - 1) + (ix - jx + WSZ - 1);
        sv += rpb[ridx * HEADS + h];
      } else {
        sv = -1e30f;                               // mask padded keys
      }
      vals[c] = sv;
      rowmax = fmaxf(rowmax, sv);
    }
    #pragma unroll
    for (int m = 1; m < 16; m <<= 1)
      rowmax = fmaxf(rowmax, __shfl_xor(rowmax, m, 32));
    float ec[4];
    float rsum = 0.f;
    #pragma unroll
    for (int c = 0; c < 4; ++c) { ec[c] = __expf(vals[c] - rowmax); rsum += ec[c]; }
    #pragma unroll
    for (int m = 1; m < 16; m <<= 1) rsum += __shfl_xor(rsum, m, 32);
    const float rinv = 1.0f / rsum;
    #pragma unroll
    for (int c = 0; c < 4; ++c)
      Ps[i * 64 + c * 16 + nl] = (_Float16)(ec[c] * rinv);
  }
  __syncthreads();

  // O = P @ V : 16x32 out per wave; K-dim = 64 padded -> 2 WMMA k-steps
  v8f o0 = {}, o1 = {};
  #pragma unroll
  for (int kk = 0; kk < 64; kk += 32) {
    const v16h pa  = load_a_frag(Ps, 64, wave * 16, kk, lane);
    const v16h vb0 = load_b_frag(Vt, 64, 0, kk, lane);
    const v16h vb1 = load_b_frag(Vt, 64, 16, kk, lane);
    o0 = __builtin_amdgcn_wmma_f32_16x16x32_f16(false, pa, false, vb0, (short)0, o0, false, false);
    o1 = __builtin_amdgcn_wmma_f32_16x16x32_f16(false, pa, false, vb1, (short)0, o1, false, false);
  }

  // scatter back token-major: attnout[t][h*32 + c]
  #pragma unroll
  for (int v = 0; v < 8; ++v) {
    const int r = wave * 16 + v + 8 * hi;
    if (r < NWT) {
      const int t = win_tok_to_global(widx, r);
      _Float16* dst = attnout + (size_t)t * DIM + h * HDIM;
      dst[nl]      = (_Float16)o0[v];
      dst[16 + nl] = (_Float16)o1[v];
    }
  }
}

// ---------------------------------------------------------------------------
extern "C" void kernel_launch(void* const* d_in, const int* in_sizes, int n_in,
                              void* d_out, int out_size, void* d_ws, size_t ws_size,
                              hipStream_t stream) {
  (void)in_sizes; (void)n_in; (void)out_size; (void)ws_size;
  const float* x      = (const float*)d_in[0];
  const float* n1w    = (const float*)d_in[1];
  const float* n1b    = (const float*)d_in[2];
  const float* qkv_w  = (const float*)d_in[3];
  const float* qkv_b  = (const float*)d_in[4];
  const float* rpb    = (const float*)d_in[5];
  const float* proj_w = (const float*)d_in[6];
  const float* proj_b = (const float*)d_in[7];
  const float* n2w    = (const float*)d_in[8];
  const float* n2b    = (const float*)d_in[9];
  const float* fc1_w  = (const float*)d_in[10];
  const float* fc1_b  = (const float*)d_in[11];
  const float* fc2_w  = (const float*)d_in[12];
  const float* fc2_b  = (const float*)d_in[13];
  float* out = (float*)d_out;

  char* ws = (char*)d_ws;
  size_t off = 0;
  auto carve = [&](size_t bytes) -> char* {
    char* p = ws + off;
    off = (off + bytes + 255) & ~(size_t)255;
    return p;
  };

  _Float16* qkvw_h = (_Float16*)carve((size_t)3 * DIM * DIM * 2);
  _Float16* projw_h = (_Float16*)carve((size_t)DIM * DIM * 2);
  _Float16* fc1w_h = (_Float16*)carve((size_t)HID * DIM * 2);
  _Float16* fc2w_h = (_Float16*)carve((size_t)DIM * HID * 2);
  _Float16* xn     = (_Float16*)carve((size_t)NTOK * DIM * 2);        // LN1(x)
  _Float16* qkvact = (_Float16*)carve((size_t)NTOK * 3 * DIM * 2);    // qkv
  _Float16* attno  = (_Float16*)carve((size_t)NTOK * DIM * 2);        // attn out
  float*    x2     = (float*)carve((size_t)NTOK * DIM * 4);           // x + proj
  _Float16* h2     = (_Float16*)carve((size_t)NTOK * DIM * 2);        // LN2(x2)
  _Float16* mlp1   = (_Float16*)carve((size_t)NTOK * HID * 2);        // gelu(fc1)

  // 1) weights -> f16 (row-major [N,K], resident in L2 thereafter)
  {
    const int n0 = 3 * DIM * DIM, n1_ = DIM * DIM, n2_ = HID * DIM, n3 = DIM * HID;
    f32_to_f16_kernel<<<(n0 + 255) / 256, 256, 0, stream>>>(qkv_w, qkvw_h, n0);
    f32_to_f16_kernel<<<(n1_ + 255) / 256, 256, 0, stream>>>(proj_w, projw_h, n1_);
    f32_to_f16_kernel<<<(n2_ + 255) / 256, 256, 0, stream>>>(fc1_w, fc1w_h, n2_);
    f32_to_f16_kernel<<<(n3 + 255) / 256, 256, 0, stream>>>(fc2_w, fc2w_h, n3);
  }

  // 2) LN1 -> f16
  layernorm_f16_kernel<<<NTOK / 8, 256, 0, stream>>>(x, n1w, n1b, xn, NTOK);

  // 3) qkv = xn @ qkv_w^T + qkv_b   (M=100352, N=576, K=192)
  {
    dim3 g(NTOK / 256, (3 * DIM) / 64);
    wmma_gemm_kernel<_Float16, EPI_NONE, false><<<g, 256, 0, stream>>>(
        xn, qkvw_h, qkv_b, nullptr, qkvact, NTOK, 3 * DIM, DIM);
  }

  // 4) shifted-window attention (gathers with shift map, scatters token-major)
  {
    dim3 g(NWIN, HEADS);
    window_attention_kernel<<<g, 128, 0, stream>>>(qkvact, rpb, attno);
  }

  // 5) x2 = x + attno @ proj_w^T + proj_b   (f32 out)
  {
    dim3 g(NTOK / 256, DIM / 64);
    wmma_gemm_kernel<float, EPI_NONE, true><<<g, 256, 0, stream>>>(
        attno, projw_h, proj_b, x, x2, NTOK, DIM, DIM);
  }

  // 6) LN2 -> f16
  layernorm_f16_kernel<<<NTOK / 8, 256, 0, stream>>>(x2, n2w, n2b, h2, NTOK);

  // 7) mlp1 = gelu(h2 @ fc1_w^T + fc1_b)   (M=100352, N=768, K=192)
  {
    dim3 g(NTOK / 256, HID / 64);
    wmma_gemm_kernel<_Float16, EPI_GELU, false><<<g, 256, 0, stream>>>(
        h2, fc1w_h, fc1_b, nullptr, mlp1, NTOK, HID, DIM);
  }

  // 8) out = x2 + mlp1 @ fc2_w^T + fc2_b   (M=100352, N=192, K=768)
  {
    dim3 g(NTOK / 256, DIM / 64);
    wmma_gemm_kernel<float, EPI_NONE, true><<<g, 256, 0, stream>>>(
        mlp1, fc2w_h, fc2_b, x2, out, NTOK, DIM, HID);
  }
}
